// AngleOrientedConv_49572512530558
// MI455X (gfx1250) — matrix operand order repaired and checked by reading, not compiled, not run on hardware
//
#include <hip/hip_runtime.h>
#include <hip/hip_fp16.h>
#include <stdint.h>

// ---------------------------------------------------------------------------
// AngleOrientedConv (4-space GAT-like attention) for gfx1250 / MI455X.
// Heavy GEMM (edge attention, [E,256]x[256,128]) runs on v_wmma_f32_16x16x32_f16.
// feat (25.6MB f32 / 12.8MB f16) is L2-resident; W per space (64KB f16) lives
// in LDS in WMMA B-fragment order.
// ---------------------------------------------------------------------------

typedef __attribute__((ext_vector_type(16))) _Float16 v16h;
typedef __attribute__((ext_vector_type(8)))  float    v8f;

#define NNODES 50000
#define FDIM   128
#define SDIM   4
#define EDGES  800000
#define KTILES 8      // K = 2F = 256 -> 8 tiles of 32
#define NTILES 8      // N = F  = 128 -> 8 tiles of 16

// --------------------------- helpers ---------------------------------------

__device__ __forceinline__ unsigned float_to_ordered(float f) {
    int b = __float_as_int(f);
    return (b >= 0) ? ((unsigned)b | 0x80000000u) : ~((unsigned)b);
}
__device__ __forceinline__ float ordered_to_float(unsigned key) {
    int b = (key & 0x80000000u) ? (int)(key & 0x7fffffffu) : (int)(~key);
    return __int_as_float(b);
}
__device__ __forceinline__ float fast_tanh(float x) {
    float t = __expf(2.0f * x);
    return (t - 1.0f) / (t + 1.0f);
}

// --------------------------- prep kernels ----------------------------------

// feat f32 -> f16 (one thread per element; 6.4M elements)
__global__ void cvt_feat_kernel(const float* __restrict__ feat,
                                _Float16* __restrict__ feat16, int n) {
    int i = blockIdx.x * blockDim.x + threadIdx.x;
    if (i < n) feat16[i] = (_Float16)feat[i];
}

// Pack W_attn [S,256,128] f32 into WMMA B-fragment order, f16.
// Flat index i = ((((k*8)+kt)*8+nt)*32 + lane)*16 + t,  total S*131072/4 = 131072
// Fragment element t of lane: K = kt*32 + 16*(lane>>4) + t,  n = nt*16 + (lane&15)
__global__ void pack_w_kernel(const float* __restrict__ W,
                              _Float16* __restrict__ wp) {
    int i = blockIdx.x * blockDim.x + threadIdx.x;
    if (i >= SDIM * KTILES * NTILES * 32 * 16) return;
    int t    =  i        & 15;
    int lane = (i >> 4)  & 31;
    int nt   = (i >> 9)  & 7;
    int kt   = (i >> 12) & 7;
    int k    =  i >> 15;
    int K = kt * 32 + ((lane >> 4) << 4) + t;
    int n = nt * 16 + (lane & 15);
    wp[i] = (_Float16)W[((long long)k * 256 + K) * 128 + n];
}

// Zero the accumulators used by atomics (must run every call).
__global__ void zero_kernel(unsigned* __restrict__ m_buf,
                            float* __restrict__ s_buf,
                            float* __restrict__ out) {
    long long i = (long long)blockIdx.x * blockDim.x + threadIdx.x;
    if (i < (long long)SDIM * NNODES) { m_buf[i] = 0u; s_buf[i] = 0.0f; }
    if (i < (long long)NNODES * SDIM * FDIM) out[i] = 0.0f;
}

// --------------------------- edge attention (WMMA) -------------------------
// grid = (E/128, S); block = 256 threads (8 waves); each wave = 16 edges.
// e[edge] = tanh([h_src||h_dst] @ W + b) . w
__global__ void
__launch_bounds__(256, 1)
edge_attn_kernel(const long long* __restrict__ src,
                 const long long* __restrict__ dst,
                 const _Float16* __restrict__ feat16,
                 const _Float16* __restrict__ wp,     // packed B fragments
                 const float* __restrict__ b_attn,
                 const float* __restrict__ w_out,
                 float* __restrict__ e_buf) {
    __shared__ uint32_t sW[KTILES * NTILES * 32 * 8];  // 64 KB: B fragments
    __shared__ float sb[FDIM];
    __shared__ float sw[FDIM];

    const int k   = blockIdx.y;
    const int tid = threadIdx.x;

    // Stage packed W for this space into LDS (4096 uint4 = 64KB, 16 per thread)
    {
        const uint4* g = ((const uint4*)wp) + (size_t)k * 4096;
        uint4* s = (uint4*)sW;
#pragma unroll
        for (int it = 0; it < 16; ++it) {
            int j = it * 256 + tid;
            s[j] = g[j];
        }
        if (tid < FDIM) {
            sb[tid] = b_attn[k * FDIM + tid];
            sw[tid] = w_out[k * FDIM + tid];
        }
    }
    __syncthreads();

    const int wave = tid >> 5;
    const int lane = tid & 31;
    const int half = lane >> 4;   // which K-half of the A/B fragment this lane holds
    const int ncol = lane & 15;

    const long long tileBase = (long long)blockIdx.x * 128 + wave * 16;
    const long long eIdx     = tileBase + (lane & 15);      // edge row M = lane&15

    const long long is = src[(long long)k * EDGES + eIdx];
    const long long id = dst[(long long)k * EDGES + eIdx];
    const _Float16* prow_s = feat16 + is * FDIM;
    const _Float16* prow_d = feat16 + id * FDIM;

    // prefetch next tile's index data (global_prefetch_b8)
    __builtin_prefetch(src + (long long)k * EDGES + eIdx + 2048, 0, 1);
    __builtin_prefetch(dst + (long long)k * EDGES + eIdx + 2048, 0, 1);

    v8f acc[NTILES];
#pragma unroll
    for (int nt = 0; nt < NTILES; ++nt)
        acc[nt] = (v8f){0.f, 0.f, 0.f, 0.f, 0.f, 0.f, 0.f, 0.f};

#pragma unroll
    for (int kt = 0; kt < KTILES; ++kt) {
        // K tiles 0..3 from h_src row, 4..7 from h_dst row (concat decomposition)
        const _Float16* base = (kt < 4) ? (prow_s + kt * 32)
                                        : (prow_d + (kt - 4) * 32);
        // A fragment (16x32 f16): lane holds K = half*16 + [0..15]
        uint4 alo = *(const uint4*)(base + half * 16);
        uint4 ahi = *(const uint4*)(base + half * 16 + 8);
        v16h a;
        ((uint4*)&a)[0] = alo;
        ((uint4*)&a)[1] = ahi;

#pragma unroll
        for (int nt = 0; nt < NTILES; ++nt) {
            const uint4* pb = (const uint4*)&sW[(((kt * 8 + nt) * 32) + lane) * 8];
            v16h b;
            ((uint4*)&b)[0] = pb[0];
            ((uint4*)&b)[1] = pb[1];
            acc[nt] = __builtin_amdgcn_wmma_f32_16x16x32_f16(
                false, a, false, b, (short)0, acc[nt], false, false);
        }
    }

    // Epilogue: e[m] = sum_n tanh(X[m][n] + b[n]) * w[n]
    // C layout: lane 0-15 vgpr j -> M=j, N=lane; lane 16-31 -> M=j+8, N=lane-16
    float esum[8];
#pragma unroll
    for (int j = 0; j < 8; ++j) esum[j] = 0.0f;

#pragma unroll
    for (int nt = 0; nt < NTILES; ++nt) {
        int n = nt * 16 + ncol;
        float bb = sb[n];
        float ww = sw[n];
#pragma unroll
        for (int j = 0; j < 8; ++j)
            esum[j] += fast_tanh(acc[nt][j] + bb) * ww;
    }

    // reduce over the 16 lanes of this half (xor masks stay within half)
#pragma unroll
    for (int off = 1; off < 16; off <<= 1) {
#pragma unroll
        for (int j = 0; j < 8; ++j)
            esum[j] += __shfl_xor(esum[j], off, 32);
    }

    if (ncol == 0) {
#pragma unroll
        for (int j = 0; j < 8; ++j)
            e_buf[(long long)k * EDGES + tileBase + half * 8 + j] = esum[j];
    }
}

// --------------------------- segment softmax -------------------------------

// pass 1: per-dst max via order-preserving uint atomicMax
__global__ void seg_max_kernel(const float* __restrict__ e_buf,
                               const long long* __restrict__ dst,
                               unsigned* __restrict__ m_buf) {
    int k = blockIdx.y;
    long long i = (long long)blockIdx.x * blockDim.x + threadIdx.x;
    if (i >= EDGES) return;
    float e = e_buf[(long long)k * EDGES + i];
    long long d = dst[(long long)k * EDGES + i];
    atomicMax(&m_buf[(long long)k * NNODES + d], float_to_ordered(e));
}

// pass 2: ex = exp(e - m[dst]); sum per dst
__global__ void seg_expsum_kernel(float* __restrict__ e_buf,
                                  const long long* __restrict__ dst,
                                  const unsigned* __restrict__ m_buf,
                                  float* __restrict__ s_buf) {
    int k = blockIdx.y;
    long long i = (long long)blockIdx.x * blockDim.x + threadIdx.x;
    if (i >= EDGES) return;
    long long d = dst[(long long)k * EDGES + i];
    float m  = ordered_to_float(m_buf[(long long)k * NNODES + d]);
    float ex = __expf(e_buf[(long long)k * EDGES + i] - m);
    e_buf[(long long)k * EDGES + i] = ex;
    atomicAdd(&s_buf[(long long)k * NNODES + d], ex);
}

// pass 3: out[dst, k*F:] += (ex/s[dst]) * feat[src]   (one wave per edge)
__global__ void scatter_kernel(const float* __restrict__ feat,
                               const long long* __restrict__ src,
                               const long long* __restrict__ dst,
                               const float* __restrict__ e_buf,
                               const float* __restrict__ s_buf,
                               float* __restrict__ out) {
    int k = blockIdx.y;
    int wave = threadIdx.x >> 5;
    int lane = threadIdx.x & 31;
    long long e = (long long)blockIdx.x * 8 + wave;
    if (e >= EDGES) return;
    long long is = src[(long long)k * EDGES + e];
    long long id = dst[(long long)k * EDGES + e];
    float alpha = e_buf[(long long)k * EDGES + e] /
                  s_buf[(long long)k * NNODES + id];
    float4 v = ((const float4*)(feat + is * FDIM))[lane];
    float* op = out + id * (SDIM * FDIM) + k * FDIM + lane * 4;
    atomicAdd(op + 0, alpha * v.x);
    atomicAdd(op + 1, alpha * v.y);
    atomicAdd(op + 2, alpha * v.z);
    atomicAdd(op + 3, alpha * v.w);
}

// --------------------------- launcher --------------------------------------

extern "C" void kernel_launch(void* const* d_in, const int* in_sizes, int n_in,
                              void* d_out, int out_size, void* d_ws, size_t ws_size,
                              hipStream_t stream) {
    const float*     feat   = (const float*)d_in[0];
    const long long* src    = (const long long*)d_in[1];
    const long long* dst    = (const long long*)d_in[2];
    const float*     W_attn = (const float*)d_in[3];
    const float*     b_attn = (const float*)d_in[4];
    const float*     w_out  = (const float*)d_in[5];
    float* out = (float*)d_out;

    // workspace carve-up (256B aligned)
    char* ws = (char*)d_ws;
    size_t off = 0;
    _Float16* feat16 = (_Float16*)(ws + off); off += (size_t)NNODES * FDIM * 2;       // 12.8 MB
    off = (off + 255) & ~(size_t)255;
    _Float16* wp     = (_Float16*)(ws + off); off += (size_t)SDIM * 32768 * 2;        // 256 KB
    off = (off + 255) & ~(size_t)255;
    float*    e_buf  = (float*)(ws + off);    off += (size_t)SDIM * EDGES * 4;        // 12.8 MB
    off = (off + 255) & ~(size_t)255;
    unsigned* m_buf  = (unsigned*)(ws + off); off += (size_t)SDIM * NNODES * 4;       // 0.8 MB
    off = (off + 255) & ~(size_t)255;
    float*    s_buf  = (float*)(ws + off);    off += (size_t)SDIM * NNODES * 4;       // 0.8 MB

    // prep
    cvt_feat_kernel<<<(NNODES * FDIM + 255) / 256, 256, 0, stream>>>(
        feat, feat16, NNODES * FDIM);
    pack_w_kernel<<<(SDIM * 32768 + 255) / 256, 256, 0, stream>>>(W_attn, wp);
    zero_kernel<<<((long long)NNODES * SDIM * FDIM + 255) / 256, 256, 0, stream>>>(
        m_buf, s_buf, out);

    // edge attention GEMM (WMMA): 128 edges / block, 4 spaces in grid.y
    dim3 gAttn(EDGES / 128, SDIM, 1);
    edge_attn_kernel<<<gAttn, 256, 0, stream>>>(src, dst, feat16, wp,
                                                b_attn, w_out, e_buf);

    // segment softmax + weighted scatter
    dim3 gE((EDGES + 255) / 256, SDIM, 1);
    seg_max_kernel<<<gE, 256, 0, stream>>>(e_buf, dst, m_buf);
    seg_expsum_kernel<<<gE, 256, 0, stream>>>(e_buf, dst, m_buf, s_buf);

    dim3 gS(EDGES / 8, SDIM, 1);
    scatter_kernel<<<gS, 256, 0, stream>>>(feat, src, dst, e_buf, s_buf, out);
}
